// Model_74955769250651
// MI455X (gfx1250) — matrix operand order repaired and checked by reading, not compiled, not run on hardware
//
#include <hip/hip_runtime.h>
#include <math.h>

// ---------------------------------------------------------------------------
// Spherical CNN forward pass for MI455X (gfx1250, wave32, WMMA).
// All dense contractions (DFT passes, spectral kernel precompute, per-degree
// mixing, MLP head) run through one generalized strided GEMM templated on
// operand mode, built on v_wmma_f32_16x16x32_f16 (f32 accumulate).
// K-dimension address tables are staged in LDS (int4 reads, division-free
// inner loop); global loads are issued in bulk phases for MLP-level MLP
// (memory-level parallelism), values masked branch-free.
// ---------------------------------------------------------------------------

#define PI_D 3.14159265358979323846

typedef _Float16 f16;
typedef __attribute__((ext_vector_type(16))) _Float16 v16h;
typedef __attribute__((ext_vector_type(8)))  float    v8f;

#define B_BATCH 2
#define R_INCH  64      // input channels
#define NA      128     // 2*B_IN
#define LMAX    32      // B_MID
#define P2      1024    // sum_{l<32} (2l+1)        = 32^2
#define P3      43680   // sum_{l<32} (2l+1)^2      = 32*63*65/3
#define VOL     262144  // 64^3
#define NPTS    2048
#define NCLS    50
#define NCATS   16
#define OCH     10      // channel chunk for 64^3 grid stages (42MB/chunk, L2-resident)
#define BIGMOD  (1<<30)

__device__ __host__ inline int off3i(int l){ return l*(2*l-1)*(2*l+1)/3; }

__device__ inline void p3dec(int p, int& l, int& m, int& n){
  int lo = 0;
  while (lo < 31 && off3i(lo+1) <= p) lo++;
  l = lo;
  int rem = p - off3i(lo);
  int W = 2*lo + 1;
  m = rem / W - lo;
  n = rem % W - lo;
}

__device__ inline void p2dec(int p, int& l, int& m){
  int lo = (int)sqrtf((float)p);
  while ((lo+1)*(lo+1) <= p) lo++;
  while (lo*lo > p) lo--;
  l = lo; m = p - lo*lo - lo;
}

__device__ inline double dipow(double x, int n){
  double r = 1.0;
  while (n > 0){ if (n & 1) r *= x; x *= x; n >>= 1; }
  return r;
}

// Wigner small-d  d^l_{mp,m}(beta), explicit log-factorial sum (f64).
__device__ double wig_d(int l, int mp, int m, double beta, const double* lf){
  double cb = cos(0.5*beta), sb = sin(0.5*beta);
  int smin = (m - mp > 0) ? (m - mp) : 0;
  int smax = (l + m < l - mp) ? (l + m) : (l - mp);
  double pref = 0.5*(lf[l+mp] + lf[l-mp] + lf[l+m] + lf[l-m]);
  double acc = 0.0;
  for (int s = smin; s <= smax; s++){
    double lt = pref - lf[l+m-s] - lf[s] - lf[mp-m+s] - lf[l-mp-s];
    double t  = exp(lt) * dipow(cb, 2*l + m - mp - 2*s) * dipow(sb, mp - m + 2*s);
    acc += ((mp - m + s) & 1) ? -t : t;
  }
  return acc;
}

// -------------------------- constant generators ----------------------------

__global__ void k_weights(float* W, int b){
  int j = blockIdx.x*blockDim.x + threadIdx.x;
  if (j >= 2*b) return;
  double s = 0.0;
  for (int k = 0; k < b; k++)
    s += sin((2.0*j+1.0)*(2.0*k+1.0)*PI_D/(4.0*b))/(2.0*k+1.0);
  W[j] = (float)((2.0/b)*sin(PI_D*(2.0*j+1.0)/(4.0*b))*s);
}

__global__ void k_betas(float* bIN, float* b32, float* bEQ){
  int i = threadIdx.x;
  if (i < 128) bIN[i] = (float)(PI_D*(2.0*i+1.0)/(4.0*64.0));
  if (i < 64)  b32[i] = (float)(PI_D*(2.0*i+1.0)/(4.0*32.0));
  if (i == 0)  bEQ[0] = (float)(PI_D*0.5);
}

__global__ void k_wigfull(const float* betas, int nb, float* out){
  __shared__ double lf[64];
  if (threadIdx.x == 0){ lf[0]=0.0; for (int i=1;i<64;i++) lf[i]=lf[i-1]+log((double)i); }
  __syncthreads();
  long long id = (long long)blockIdx.x*blockDim.x + threadIdx.x;
  if (id >= (long long)nb*P3) return;
  int t = (int)(id / P3), p = (int)(id % P3);
  int l,m,n; p3dec(p,l,m,n);
  out[id] = (float)wig_d(l, m, n, (double)betas[t], lf);
}

__global__ void k_wigcol0(const float* betas, int nb, float* out){
  __shared__ double lf[64];
  if (threadIdx.x == 0){ lf[0]=0.0; for (int i=1;i<64;i++) lf[i]=lf[i-1]+log((double)i); }
  __syncthreads();
  long long id = (long long)blockIdx.x*blockDim.x + threadIdx.x;
  if (id >= (long long)nb*P2) return;
  int t = (int)(id / P2), p = (int)(id % P2);
  int l,m; p2dec(p,l,m);
  out[id] = (float)wig_d(l, m, 0, (double)betas[t], lf);
}

__global__ void k_dft(float2* out, int n, float sgn){
  int id = blockIdx.x*blockDim.x + threadIdx.x;
  if (id >= n*n) return;
  int j = id / n, k = id % n;
  double th = (double)sgn * 2.0 * PI_D * (double)(j*k) / (double)n;
  out[id] = make_float2((float)cos(th), (float)sin(th));
}

// S2Kp[i][p2=(l,m)] = exp(-i*m*alpha_i) * d^l_{m,0}(pi/2),  i<128
__global__ void k_s2k(const float* DEQ, float2* S2Kp){
  int id = blockIdx.x*blockDim.x + threadIdx.x;
  if (id >= 128*P2) return;
  int i = id / P2, p = id % P2;
  int l,m; p2dec(p,l,m);
  float d = DEQ[off3i(l) + (m+l)*(2*l+1) + l];
  double ang = -(double)m * (2.0*PI_D*(double)i/128.0);
  S2Kp[id] = make_float2((float)(cos(ang)*d), (float)(sin(ang)*d));
}

// SO3Kp[i][p3=(l,m,n)] = exp(-i*m*alpha_i) * d^l_{m,n}(pi/2),  i<64
__global__ void k_so3k(const float* DEQ, float2* SO3Kp){
  long long id = (long long)blockIdx.x*blockDim.x + threadIdx.x;
  if (id >= (long long)64*P3) return;
  int i = (int)(id / P3), p = (int)(id % P3);
  int l,m,n; p3dec(p,l,m,n);
  float d = DEQ[p];
  double ang = -(double)m * (2.0*PI_D*(double)i/64.0);
  SO3Kp[id] = make_float2((float)(cos(ang)*d), (float)(sin(ang)*d));
}

// ------------------------ generalized WMMA GEMM ----------------------------
// C[r,c] (+batch) = sum_k A[r,k] * B[k,c].
// MODE: 0 = complex*complex->complex, 1 = real*complex->complex,
//       2 = real*real->real.
// Addressing: idx = off + (r/RM)*s1 + (r%RM)*s0 + koff[k]
// (strides in elements: float if real, float2 if complex).
// koffA/koffB built once per block in LDS; fetched per k-step as int4 runs.

struct GemmP {
  const float* A; const float* B; float* C;
  int M, N, K;
  int ARM, AKM, BKM, BNM, CRM, CNM;
  long long Ar1, Ar0, Ak1, Ak0, Aoff;
  long long Bk1, Bk0, Bn1, Bn0, Boff;
  long long Cr1, Cr0, Cn1, Cn0, Coff;
  long long Abat, Bbat, Cbat;
};

__device__ inline v8f wmma16(v16h a, v16h b, v8f c){
  return __builtin_amdgcn_wmma_f32_16x16x32_f16(false, a, false, b, (short)0, c, false, false);
}

template<int MODE>
__global__ void __launch_bounds__(256) gemm_wmma(GemmP p){
  extern __shared__ int koff[];
  const int Kpad = (p.K + 31) & ~31;
  int* koffA = koff;
  int* koffB = koff + Kpad;
  for (int k = threadIdx.x; k < Kpad; k += 256){
    koffA[k] = (k < p.K) ? (int)((long long)(k / p.AKM)*p.Ak1 + (long long)(k % p.AKM)*p.Ak0) : 0;
    koffB[k] = (k < p.K) ? (int)((long long)(k / p.BKM)*p.Bk1 + (long long)(k % p.BKM)*p.Bk0) : 0;
  }
  __syncthreads();

  const int lane = threadIdx.x & 31;
  const int wave = threadIdx.x >> 5;             // 8 waves: 2 (M) x 4 (N) tiles
  const int tm = (blockIdx.y*2 + (wave>>2))*16;
  const int tn = (blockIdx.x*4 + (wave&3))*16;
  const int l15 = lane & 15;
  const bool hi = lane >= 16;
  const int ha = hi ? 8 : 0;
  const int hb = hi ? 16 : 0;

  const int row = tm + l15;
  const int col = tn + l15;
  const bool rok = row < p.M;
  const bool nok = col < p.N;
  const int rowc = rok ? row : 0;                // clamp -> loads always in bounds
  const int colc = nok ? col : 0;

  const long long abase = (long long)blockIdx.z*p.Abat + p.Aoff
      + (long long)(rowc / p.ARM)*p.Ar1 + (long long)(rowc % p.ARM)*p.Ar0;
  const long long bbase = (long long)blockIdx.z*p.Bbat + p.Boff
      + (long long)(colc / p.BNM)*p.Bn1 + (long long)(colc % p.BNM)*p.Bn0;

  const float2* A2 = (const float2*)p.A;
  const float2* B2 = (const float2*)p.B;

  v8f cre = {0.f,0.f,0.f,0.f,0.f,0.f,0.f,0.f};
  v8f cim = cre;

  for (int k0 = 0; k0 < p.K; k0 += 32){
    // phase 1: fetch all k-offsets from LDS as contiguous int4 runs
    const int4* kA1 = (const int4*)(koffA + k0 + hb);        // A elems 0..7
    const int4* kA2 = (const int4*)(koffA + k0 + hb + 8);    // A elems 8..15
    const int4* kB  = (const int4*)(koffB + k0 + hb);        // B elems 0..15
    const int4 a0 = kA1[0], a1 = kA1[1], a2 = kA2[0], a3 = kA2[1];
    const int4 b0 = kB[0],  b1 = kB[1],  b2 = kB[2],  b3 = kB[3];
    const int offA[16] = {a0.x,a0.y,a0.z,a0.w, a1.x,a1.y,a1.z,a1.w,
                          a2.x,a2.y,a2.z,a2.w, a3.x,a3.y,a3.z,a3.w};
    const int offB[16] = {b0.x,b0.y,b0.z,b0.w, b1.x,b1.y,b1.z,b1.w,
                          b2.x,b2.y,b2.z,b2.w, b3.x,b3.y,b3.z,b3.w};

    // phase 2: bulk global loads (no per-element waits)
    float2 va2[16]; float va1[16];
    #pragma unroll
    for (int e = 0; e < 16; e++){
      const long long ia = abase + offA[e];
      if (MODE == 0) va2[e] = A2[ia]; else va1[e] = p.A[ia];
    }
    float2 vb2[16]; float vb1[16];
    #pragma unroll
    for (int e = 0; e < 16; e++){
      const long long ib = bbase + offB[e];
      if (MODE == 2) vb1[e] = p.B[ib]; else vb2[e] = B2[ib];
    }

    // phase 3: mask + convert into WMMA fragments
    v16h ar, aim, ain, br, bim;
    #pragma unroll
    for (int e = 0; e < 16; e++){
      const int ka = k0 + hb + e;                            // A 16x32 layout, contiguous k (matches B)
      const bool okA = rok && (ka < p.K);
      if (MODE == 0){
        float arv = okA ? va2[e].x : 0.f;
        float aiv = okA ? va2[e].y : 0.f;
        ar[e]  = (f16)arv;
        aim[e] = (f16)aiv;
        ain[e] = (f16)(-aiv);                                // no A-NEG on 16-bit WMMA
      } else {
        ar[e] = (f16)(okA ? va1[e] : 0.f);
      }
      const int kb = k0 + hb + e;                            // B 32x16 layout
      const bool okB = nok && (kb < p.K);
      if (MODE == 2){
        br[e] = (f16)(okB ? vb1[e] : 0.f);
      } else {
        br[e]  = (f16)(okB ? vb2[e].x : 0.f);
        bim[e] = (f16)(okB ? vb2[e].y : 0.f);
      }
    }

    // phase 4: matrix ops
    cre = wmma16(ar, br, cre);
    if (MODE == 0){
      cre = wmma16(ain, bim, cre);
      cim = wmma16(ar,  bim, cim);
      cim = wmma16(aim, br,  cim);
    } else if (MODE == 1){
      cim = wmma16(ar, bim, cim);
    }
  }

  float2* C2 = (float2*)p.C;
  #pragma unroll
  for (int r = 0; r < 8; r++){
    const int rr = tm + r + (hi ? 8 : 0);        // ISA f32 C/D layout
    const int cc = tn + l15;
    if (rr < p.M && cc < p.N){
      const long long idx = (long long)blockIdx.z*p.Cbat + p.Coff
        + (long long)(rr / p.CRM)*p.Cr1 + (long long)(rr % p.CRM)*p.Cr0
        + (long long)(cc / p.CNM)*p.Cn1 + (long long)(cc % p.CNM)*p.Cn0;
      if (MODE == 2) p.C[idx] = cre[r];
      else { float2 t; t.x = cre[r]; t.y = cim[r]; C2[idx] = t; }
    }
  }
}

// --------------------------- pipeline kernels ------------------------------

// cs1[b,f,(l,m)] = sum_j W64[j]*DIN0[j,(l,m)]*xf[b,f,j, m mod 128]
__global__ void k_cs1(const float2* xf, const float* DIN0, const float* W64, float2* cs){
  int id = blockIdx.x*blockDim.x + threadIdx.x;
  if (id >= B_BATCH*R_INCH*P2) return;
  int p = id % P2; int bf = id / P2;
  int l,m; p2dec(p,l,m);
  int colk = m & 127;
  const float2* xr = xf + (long long)bf*NA*NA;
  float sr = 0.f, si = 0.f;
  for (int j = 0; j < NA; j++){
    float w = W64[j]*DIN0[j*P2 + p];
    float2 v = xr[j*NA + colk];
    sr += w*v.x; si += w*v.y;
  }
  cs[id] = make_float2(sr, si);
}

// Fm[b,oc,j,mt,nt] = sum_l (2l+1) d^l_{m,n}(beta_j) zs[b,o,(l,m,n)]
__global__ void k_synth(const float2* zs, const float* D32, float2* Fm, int Fout, int o0){
  long long id = (long long)blockIdx.x*blockDim.x + threadIdx.x;
  if (id >= (long long)B_BATCH*OCH*VOL) return;
  int nt = (int)(id & 63), mt = (int)((id>>6) & 63), j = (int)((id>>12) & 63);
  int v3 = (int)(id >> 18); int oc = v3 % OCH; int b = v3 / OCH;
  int m = (mt <= 31) ? mt : mt - 64;
  int n = (nt <= 31) ? nt : nt - 64;
  int am = m < 0 ? -m : m, an = n < 0 ? -n : n;
  int lmin = am > an ? am : an;
  float sr = 0.f, si = 0.f;
  if (lmin <= 31){
    const long long zb = ((long long)b*Fout + (o0+oc))*P3;
    for (int l = lmin; l < 32; l++){
      int W = 2*l + 1;
      int o3 = off3i(l) + (m+l)*W + (n+l);
      float d = (float)(2*l+1) * D32[(long long)j*P3 + o3];
      float2 z = zs[zb + o3];
      sr += d*z.x; si += d*z.y;
    }
  }
  Fm[id] = make_float2(sr, si);
}

__global__ void k_realbias(const float2* Fm, const float* cb, float* h, int Fout, int o0){
  long long id = (long long)blockIdx.x*blockDim.x + threadIdx.x;
  if (id >= (long long)B_BATCH*OCH*VOL) return;
  int v = (int)(id % VOL); int rest = (int)(id / VOL);
  int oc = rest % OCH, b = rest / OCH;
  h[((long long)b*Fout + (o0+oc))*VOL + v] = Fm[id].x + cb[o0+oc];
}

__global__ void k_bnstats(const float* h, int F, float* stats){
  __shared__ double ss[256], sq[256];
  int f = blockIdx.x;
  double s = 0.0, q = 0.0;
  for (long long i = threadIdx.x; i < (long long)B_BATCH*VOL; i += blockDim.x){
    long long b = i / VOL, v = i % VOL;
    float x = h[(b*F + f)*(long long)VOL + v];
    s += x; q += (double)x*x;
  }
  ss[threadIdx.x] = s; sq[threadIdx.x] = q;
  __syncthreads();
  for (int st = 128; st > 0; st >>= 1){
    if (threadIdx.x < st){ ss[threadIdx.x] += ss[threadIdx.x+st]; sq[threadIdx.x] += sq[threadIdx.x+st]; }
    __syncthreads();
  }
  if (threadIdx.x == 0){
    double cnt = (double)B_BATCH*VOL;
    double mu = ss[0]/cnt;
    double var = sq[0]/cnt - mu*mu;
    stats[2*f] = (float)mu; stats[2*f+1] = (float)var;
  }
}

__global__ void k_bnrelu(float* h, int F, const float* stats, const float* g, const float* be){
  long long id = (long long)blockIdx.x*blockDim.x + threadIdx.x;
  if (id >= (long long)B_BATCH*F*VOL) return;
  int f = (int)((id / VOL) % F);
  float x = h[id];
  float y = g[f]*(x - stats[2*f])*rsqrtf(stats[2*f+1] + 1e-5f) + be[f];
  h[id] = y > 0.f ? y : 0.f;
}

// cs2[b,f0+fc,(l,m,n)] = sum_j W32[j]*d^l_{m,n}(beta_j)*X2[b,fc,j, m&63, n&63]
__global__ void k_cs2(const float2* X2, const float* D32, const float* W32,
                      float2* cs2, int F, int f0){
  long long id = (long long)blockIdx.x*blockDim.x + threadIdx.x;
  if (id >= (long long)B_BATCH*OCH*P3) return;
  int p = (int)(id % P3); int rest = (int)(id / P3);
  int fc = rest % OCH, b = rest / OCH;
  int l,m,n; p3dec(p,l,m,n);
  int cm = m & 63, cn = n & 63;
  const float2* xb = X2 + (long long)(b*OCH + fc)*VOL;
  float sr = 0.f, si = 0.f;
  for (int j = 0; j < 64; j++){
    float w = W32[j]*D32[(long long)j*P3 + p];
    float2 v = xb[((long long)j << 12) + (cm << 6) + cn];
    sr += w*v.x; si += w*v.y;
  }
  cs2[((long long)b*F + (f0+fc))*P3 + p] = make_float2(sr, si);
}

__global__ void k_gridsample(const float* h, const float* grid, float* feats){
  int id = blockIdx.x*blockDim.x + threadIdx.x;
  if (id >= B_BATCH*NPTS*NCLS) return;
  int c = id % NCLS; int pt = (id/NCLS) % NPTS; int b = id/(NCLS*NPTS);
  const float* gp = grid + ((long long)b*NPTS + pt)*3;
  float ix = ((gp[0] + 1.f)*64.f - 1.f)*0.5f;
  float iy = ((gp[1] + 1.f)*64.f - 1.f)*0.5f;
  float iz = ((gp[2] + 1.f)*64.f - 1.f)*0.5f;
  float x0 = floorf(ix), y0 = floorf(iy), z0 = floorf(iz);
  const float* hb = h + ((long long)b*NCLS + c)*VOL;
  float out = 0.f;
  #pragma unroll
  for (int dz = 0; dz < 2; dz++)
  #pragma unroll
  for (int dy = 0; dy < 2; dy++)
  #pragma unroll
  for (int dx = 0; dx < 2; dx++){
    float xc = x0 + dx, yc = y0 + dy, zc = z0 + dz;
    float wx = dx ? (ix - x0) : (x0 + 1.f - ix);
    float wy = dy ? (iy - y0) : (y0 + 1.f - iy);
    float wz = dz ? (iz - z0) : (z0 + 1.f - iz);
    bool valid = xc >= 0.f && xc < 64.f && yc >= 0.f && yc < 64.f && zc >= 0.f && zc < 64.f;
    int xi = (int)fminf(fmaxf(xc, 0.f), 63.f);
    int yi = (int)fminf(fmaxf(yc, 0.f), 63.f);
    int zi = (int)fminf(fmaxf(zc, 0.f), 63.f);
    float v = hb[(zi << 12) + (yi << 6) + xi];
    out += v * (valid ? wx*wy*wz : 0.f);
  }
  feats[id] = out;
}

__global__ void k_concat(const float* feats, const float* onehot, float* P){
  int id = blockIdx.x*blockDim.x + threadIdx.x;
  if (id >= B_BATCH*NPTS*(NCLS+NCATS)) return;
  int jc = id % (NCLS+NCATS); int row = id / (NCLS+NCATS);
  int b = row / NPTS;
  P[id] = (jc < NCLS) ? feats[(long long)row*NCLS + jc] : onehot[b*NCATS + (jc - NCLS)];
}

__global__ void k_biasrelu(float* X, const float* bias, int n, int cols){
  int id = blockIdx.x*blockDim.x + threadIdx.x;
  if (id >= n) return;
  float v = X[id] + bias[id % cols];
  X[id] = v > 0.f ? v : 0.f;
}

__global__ void k_logsoftmax(const float* H2, const float* lb2, float* out){
  int row = blockIdx.x*blockDim.x + threadIdx.x;
  if (row >= B_BATCH*NPTS) return;
  float v[NCLS];
  float mx = -1e30f;
  for (int c = 0; c < NCLS; c++){ v[c] = H2[row*NCLS + c] + lb2[c]; mx = v[c] > mx ? v[c] : mx; }
  float s = 0.f;
  for (int c = 0; c < NCLS; c++) s += expf(v[c] - mx);
  float lse = mx + logf(s);
  for (int c = 0; c < NCLS; c++) out[row*NCLS + c] = v[c] - lse;
}

// ------------------------------- host side ---------------------------------

static GemmP gz(){
  GemmP p{};
  p.ARM = p.AKM = p.BKM = p.BNM = p.CRM = p.CNM = BIGMOD;
  return p;
}

static void run_gemm(const GemmP& p, int batch, int mode, hipStream_t s){
  int Kpad = (p.K + 31) & ~31;
  size_t sh = (size_t)2*Kpad*4;
  dim3 g((unsigned)((p.N + 63)/64), (unsigned)((p.M + 31)/32), (unsigned)batch);
  if (mode == 0)      gemm_wmma<0><<<g, 256, sh, s>>>(p);
  else if (mode == 1) gemm_wmma<1><<<g, 256, sh, s>>>(p);
  else                gemm_wmma<2><<<g, 256, sh, s>>>(p);
}

// inverse SO(3) FFT:  zs -> h (chunked over output channels)
static void ifft_stage(const float2* zs, const float* D32, const float2* E64i,
                       float2* FmC, float2* G1C, const float* cb, float* h,
                       int Fout, hipStream_t stream){
  long long n = (long long)B_BATCH*OCH*VOL;
  for (int o0 = 0; o0 < Fout; o0 += OCH){
    k_synth<<<(unsigned)((n+255)/256), 256, 0, stream>>>(zs, D32, FmC, Fout, o0);
    GemmP p1 = gz();                 // contract n-tilde
    p1.A = (const float*)FmC; p1.B = (const float*)E64i; p1.C = (float*)G1C;
    p1.M = B_BATCH*OCH*64*64; p1.K = 64; p1.N = 64;
    p1.Ar0 = 64; p1.Ak0 = 1; p1.Bk0 = 64; p1.Bn0 = 1; p1.Cr0 = 64; p1.Cn0 = 1;
    run_gemm(p1, 1, 0, stream);
    GemmP p2 = gz();                 // contract m-tilde (batched slabs, E symmetric)
    p2.A = (const float*)E64i; p2.B = (const float*)G1C; p2.C = (float*)FmC;
    p2.M = 64; p2.K = 64; p2.N = 64;
    p2.Ar0 = 64; p2.Ak0 = 1;
    p2.Bk0 = 64; p2.Bn0 = 1; p2.Bbat = 4096;
    p2.Cr0 = 64; p2.Cn0 = 1; p2.Cbat = 4096;
    run_gemm(p2, B_BATCH*OCH*64, 0, stream);
    k_realbias<<<(unsigned)((n+255)/256), 256, 0, stream>>>(FmC, cb, h, Fout, o0);
  }
}

static void bn_stage(float* h, int F, float* stats, const float* g, const float* be,
                     hipStream_t stream){
  k_bnstats<<<F, 256, 0, stream>>>(h, F, stats);
  long long n = (long long)B_BATCH*F*VOL;
  k_bnrelu<<<(unsigned)((n+255)/256), 256, 0, stream>>>(h, F, stats, g, be);
}

// forward SO(3) FFT + coefficient extraction:  h -> cs2 (chunked over channels)
static void fwd_stage(const float* h, const float2* E64f, const float* D32, const float* W32,
                      float2* FmC, float2* G1C, float2* cs2, int F, hipStream_t stream){
  for (int f0 = 0; f0 < F; f0 += OCH){
    GemmP p1 = gz();                 // contract a2 (A is real h)
    p1.A = h; p1.B = (const float*)E64f; p1.C = (float*)FmC;
    p1.M = B_BATCH*OCH*64*64; p1.K = 64; p1.N = 64;
    p1.ARM = OCH*4096; p1.Ar1 = (long long)F*VOL; p1.Ar0 = 64; p1.Aoff = (long long)f0*VOL;
    p1.Ak0 = 1; p1.Bk0 = 64; p1.Bn0 = 1; p1.Cr0 = 64; p1.Cn0 = 1;
    run_gemm(p1, 1, 1, stream);
    GemmP p2 = gz();                 // contract a1 (batched slabs)
    p2.A = (const float*)E64f; p2.B = (const float*)FmC; p2.C = (float*)G1C;
    p2.M = 64; p2.K = 64; p2.N = 64;
    p2.Ar0 = 64; p2.Ak0 = 1;
    p2.Bk0 = 64; p2.Bn0 = 1; p2.Bbat = 4096;
    p2.Cr0 = 64; p2.Cn0 = 1; p2.Cbat = 4096;
    run_gemm(p2, B_BATCH*OCH*64, 0, stream);
    long long n = (long long)B_BATCH*OCH*P3;
    k_cs2<<<(unsigned)((n+255)/256), 256, 0, stream>>>(G1C, D32, W32, cs2, F, f0);
  }
}

// per-degree spectral mixing for SO(3) layers: kml_l = k (x) SO3K_l ; zs_l = cs2 x kml_l
static void mix_stage(const float2* cs2, const float* kmat, const float2* SO3Kp,
                      float2* kml, float2* zs, int Fin, int Fout, hipStream_t stream){
  for (int l = 0; l < 32; l++){
    int W = 2*l + 1; long long o3 = off3i(l);
    {
      GemmP p = gz();
      p.A = kmat; p.B = (const float*)SO3Kp; p.C = (float*)kml;
      p.M = Fin*Fout; p.K = 64; p.N = W*W;
      p.ARM = Fout; p.Ar1 = (long long)Fout*64; p.Ar0 = 64; p.Ak0 = 1;
      p.Bk0 = P3; p.BNM = W; p.Bn1 = W; p.Bn0 = 1; p.Boff = o3;
      p.CRM = Fout; p.Cr1 = (long long)Fout*W*W; p.Cr0 = (long long)W*W;
      p.CNM = W; p.Cn1 = W; p.Cn0 = 1;
      run_gemm(p, 1, 1, stream);
    }
    {
      GemmP p = gz();
      p.A = (const float*)cs2; p.B = (const float*)kml; p.C = (float*)zs;
      p.M = B_BATCH*W; p.K = Fin*W; p.N = Fout*W;
      p.ARM = W; p.Ar1 = (long long)Fin*P3; p.Ar0 = W; p.Aoff = o3;
      p.AKM = W; p.Ak1 = P3; p.Ak0 = 1;
      p.BKM = W; p.Bk1 = (long long)Fout*W*W; p.Bk0 = W;
      p.BNM = W; p.Bn1 = (long long)W*W; p.Bn0 = 1;
      p.CRM = W; p.Cr1 = (long long)Fout*P3; p.Cr0 = W;
      p.CNM = W; p.Cn1 = P3; p.Cn0 = 1; p.Coff = o3;
      run_gemm(p, 1, 0, stream);
    }
  }
}

extern "C" void kernel_launch(void* const* d_in, const int* in_sizes, int n_in,
                              void* d_out, int out_size, void* d_ws, size_t ws_size,
                              hipStream_t stream){
  (void)in_sizes; (void)n_in; (void)out_size; (void)ws_size;
  const float* x    = (const float*)d_in[0];
  const float* tgt  = (const float*)d_in[1];
  const float* oneh = (const float*)d_in[2];
  const float* k1   = (const float*)d_in[3];
  const float* cb1  = (const float*)d_in[4];
  const float* g1   = (const float*)d_in[5];
  const float* be1  = (const float*)d_in[6];
  const float* k2   = (const float*)d_in[7];
  const float* cb2  = (const float*)d_in[8];
  const float* g2   = (const float*)d_in[9];
  const float* be2  = (const float*)d_in[10];
  const float* k3   = (const float*)d_in[11];
  const float* cb3  = (const float*)d_in[12];
  const float* g3   = (const float*)d_in[13];
  const float* be3  = (const float*)d_in[14];
  const float* lw1  = (const float*)d_in[15];
  const float* lb1  = (const float*)d_in[16];
  const float* lw2  = (const float*)d_in[17];
  const float* lb2  = (const float*)d_in[18];
  float* out = (float*)d_out;

  char* base = (char*)d_ws; size_t off = 0;
  auto alloc = [&](size_t bytes)->void*{
    size_t cur = (off + 255) & ~(size_t)255; off = cur + bytes; return base + cur;
  };

  float*  W64   = (float*) alloc(128*4);
  float*  W32   = (float*) alloc(64*4);
  float*  bIN   = (float*) alloc(128*4);
  float*  b32   = (float*) alloc(64*4);
  float*  bEQ   = (float*) alloc(4);
  float*  DEQ   = (float*) alloc((size_t)P3*4);
  float*  DIN0  = (float*) alloc((size_t)128*P2*4);
  float*  D32   = (float*) alloc((size_t)64*P3*4);
  float2* S2Kp  = (float2*)alloc((size_t)128*P2*8);
  float2* SO3Kp = (float2*)alloc((size_t)64*P3*8);
  float2* F128  = (float2*)alloc((size_t)128*128*8);
  float2* E64f  = (float2*)alloc((size_t)64*64*8);
  float2* E64i  = (float2*)alloc((size_t)64*64*8);
  float2* km1   = (float2*)alloc((size_t)64*40*P2*8);
  float2* xf    = (float2*)alloc((size_t)B_BATCH*R_INCH*NA*NA*8);
  float2* cs1   = (float2*)alloc((size_t)B_BATCH*R_INCH*P2*8);
  float2* cs2   = (float2*)alloc((size_t)B_BATCH*40*P3*8);
  float2* zs    = (float2*)alloc((size_t)B_BATCH*50*P3*8);
  float2* FmC   = (float2*)alloc((size_t)B_BATCH*OCH*VOL*8);
  float2* G1C   = (float2*)alloc((size_t)B_BATCH*OCH*VOL*8);
  float2* kml   = (float2*)alloc((size_t)40*50*63*63*8);
  float*  h     = (float*) alloc((size_t)B_BATCH*50*VOL*4);
  float*  stats = (float*) alloc(100*4);
  float*  feats = (float*) alloc((size_t)B_BATCH*NPTS*NCLS*4);
  float*  Pbuf  = (float*) alloc((size_t)B_BATCH*NPTS*(NCLS+NCATS)*4);
  float*  H1    = (float*) alloc((size_t)B_BATCH*NPTS*NCLS*4);
  float*  H2    = (float*) alloc((size_t)B_BATCH*NPTS*NCLS*4);

  // ---- constants (regenerated each launch; deterministic) ----
  k_weights<<<1, 128, 0, stream>>>(W64, 64);
  k_weights<<<1, 64, 0, stream>>>(W32, 32);
  k_betas<<<1, 128, 0, stream>>>(bIN, b32, bEQ);
  { long long n = (long long)64*P3;
    k_wigfull<<<(unsigned)((n+255)/256), 256, 0, stream>>>(b32, 64, D32); }
  { long long n = P3;
    k_wigfull<<<(unsigned)((n+255)/256), 256, 0, stream>>>(bEQ, 1, DEQ); }
  { long long n = (long long)128*P2;
    k_wigcol0<<<(unsigned)((n+255)/256), 256, 0, stream>>>(bIN, 128, DIN0); }
  k_s2k<<<(128*P2+255)/256, 256, 0, stream>>>(DEQ, S2Kp);
  { long long n = (long long)64*P3;
    k_so3k<<<(unsigned)((n+255)/256), 256, 0, stream>>>(DEQ, SO3Kp); }
  k_dft<<<(128*128+255)/256, 256, 0, stream>>>(F128, 128, -1.f);
  k_dft<<<(64*64+255)/256, 256, 0, stream>>>(E64f, 64, -1.f);
  k_dft<<<(64*64+255)/256, 256, 0, stream>>>(E64i, 64,  1.f);

  // km1[f,o,(l,n)] = sum_i k1[f,o,i] * S2Kp[i,(l,n)]   (WMMA, A real)
  { GemmP p = gz();
    p.A = k1; p.B = (const float*)S2Kp; p.C = (float*)km1;
    p.M = 64*40; p.K = 128; p.N = P2;
    p.ARM = 40; p.Ar1 = (long long)40*128; p.Ar0 = 128; p.Ak0 = 1;
    p.Bk0 = P2; p.Bn0 = 1;
    p.CRM = 40; p.Cr1 = (long long)40*P2; p.Cr0 = P2; p.Cn0 = 1;
    run_gemm(p, 1, 1, stream);
  }

  // ---- S^2 FFT along phi: xf = x @ DFT128 (WMMA, A real) ----
  { GemmP p = gz();
    p.A = x; p.B = (const float*)F128; p.C = (float*)xf;
    p.M = B_BATCH*R_INCH*NA; p.K = NA; p.N = NA;
    p.Ar0 = NA; p.Ak0 = 1; p.Bk0 = NA; p.Bn0 = 1; p.Cr0 = NA; p.Cn0 = 1;
    run_gemm(p, 1, 1, stream);
  }
  k_cs1<<<(B_BATCH*R_INCH*P2+255)/256, 256, 0, stream>>>(xf, DIN0, W64, cs1);

  // ---- layer 1 spectral mixing (per-degree WMMA GEMMs) ----
  for (int l = 0; l < 32; l++){
    int W = 2*l + 1;
    GemmP p = gz();
    p.A = (const float*)cs1; p.B = (const float*)km1; p.C = (float*)zs;
    p.M = B_BATCH*W; p.K = R_INCH; p.N = 40*W;
    p.ARM = W; p.Ar1 = (long long)R_INCH*P2; p.Ar0 = 1; p.Aoff = (long long)l*l;
    p.Ak0 = P2;
    p.Bk0 = (long long)40*P2;
    p.BNM = W; p.Bn1 = P2; p.Bn0 = 1; p.Boff = (long long)l*l;
    p.CRM = W; p.Cr1 = (long long)40*P3; p.Cr0 = W;
    p.CNM = W; p.Cn1 = P3; p.Cn0 = 1; p.Coff = off3i(l);
    run_gemm(p, 1, 0, stream);
  }
  ifft_stage(zs, D32, E64i, FmC, G1C, cb1, h, 40, stream);
  bn_stage(h, 40, stats, g1, be1, stream);

  // ---- layer 2 ----
  fwd_stage(h, E64f, D32, W32, FmC, G1C, cs2, 40, stream);
  mix_stage(cs2, k2, SO3Kp, kml, zs, 40, 40, stream);
  ifft_stage(zs, D32, E64i, FmC, G1C, cb2, h, 40, stream);
  bn_stage(h, 40, stats, g2, be2, stream);

  // ---- layer 3 ----
  fwd_stage(h, E64f, D32, W32, FmC, G1C, cs2, 40, stream);
  mix_stage(cs2, k3, SO3Kp, kml, zs, 40, 50, stream);
  ifft_stage(zs, D32, E64i, FmC, G1C, cb3, h, 50, stream);
  bn_stage(h, 50, stats, g3, be3, stream);

  // ---- head: grid sample + concat + MLP (WMMA) + log-softmax ----
  k_gridsample<<<(B_BATCH*NPTS*NCLS+255)/256, 256, 0, stream>>>(h, tgt, feats);
  k_concat<<<(B_BATCH*NPTS*(NCLS+NCATS)+255)/256, 256, 0, stream>>>(feats, oneh, Pbuf);
  { GemmP p = gz();
    p.A = Pbuf; p.B = lw1; p.C = H1;
    p.M = B_BATCH*NPTS; p.K = NCLS+NCATS; p.N = NCLS;
    p.Ar0 = NCLS+NCATS; p.Ak0 = 1; p.Bk0 = NCLS; p.Bn0 = 1; p.Cr0 = NCLS; p.Cn0 = 1;
    run_gemm(p, 1, 2, stream);
  }
  k_biasrelu<<<(B_BATCH*NPTS*NCLS+255)/256, 256, 0, stream>>>(H1, lb1, B_BATCH*NPTS*NCLS, NCLS);
  { GemmP p = gz();
    p.A = H1; p.B = lw2; p.C = H2;
    p.M = B_BATCH*NPTS; p.K = NCLS; p.N = NCLS;
    p.Ar0 = NCLS; p.Ak0 = 1; p.Bk0 = NCLS; p.Bn0 = 1; p.Cr0 = NCLS; p.Cn0 = 1;
    run_gemm(p, 1, 2, stream);
  }
  k_logsoftmax<<<(B_BATCH*NPTS+255)/256, 256, 0, stream>>>(H2, lb2, out);
}
